// HMLSTM_42442866819315
// MI455X (gfx1250) — matrix operand order repaired and checked by reading, not compiled, not run on hardware
//
#include <hip/hip_runtime.h>

// HMLSTM forward for MI455X (gfx1250): bf16 WMMA GEMMs + fp32 cell math.
// L=3, H=1024, IN=1024, T=128, B=128.
// Weights (~69 MB bf16) are L2-resident (192 MB L2); GEMM tiling minimizes
// redundant weight (B-matrix) traffic: 2(M)x4(N) wave grid per 256-thread block.
// GEMM inner loop is software-pipelined (register double-buffer) so WMMA issue
// overlaps the next iteration's global loads instead of s_wait_loadcnt 0x0 stalls.

typedef __attribute__((ext_vector_type(16))) __bf16 v16bf;
typedef __attribute__((ext_vector_type(8)))  __bf16 v8bf;
typedef __attribute__((ext_vector_type(8)))  float  v8f;

__device__ __forceinline__ float sigf(float x) { return 1.0f / (1.0f + expf(-x)); }

// A-fragment (16x32 bf16): per lane, K elements {0..7} and {16..23} relative to its 8-chunk.
__device__ __forceinline__ v16bf ld16bf_split8(const __bf16* p) {
  v8bf lo = *(const v8bf*)p;
  v8bf hi = *(const v8bf*)(p + 16);
  return __builtin_shufflevector(lo, hi, 0,1,2,3,4,5,6,7,8,9,10,11,12,13,14,15);
}
// Unused: per lane, 16 contiguous K elements; B fragments are loaded with ld16bf_split8 like A.
__device__ __forceinline__ v16bf ld16bf_cont(const __bf16* p) {
  v8bf lo = *(const v8bf*)p;
  v8bf hi = *(const v8bf*)(p + 8);
  return __builtin_shufflevector(lo, hi, 0,1,2,3,4,5,6,7,8,9,10,11,12,13,14,15);
}

struct Frags { v16bf b, a0, a1, a2, a3; };

// C = A(MxK bf16 row-major) * B + bias, B given transposed (Bt: N x K row-major bf16).
// C: MxN fp32 row-major. Block: 256 threads (8 waves), block tile 128(M) x 64(N).
// Wave (wm, wn): wm in {0,1} covers rows wm*64..+63 (4 WMMA M-tiles), wn in {0..3}
// covers one 16-wide N-tile. B fragments loaded 2x per block; A redundancy lands
// on the small cache-hot activation matrix.
__global__ __launch_bounds__(256)
void gemm_bf16_wmma(const __bf16* __restrict__ A, const __bf16* __restrict__ Bt,
                    const float* __restrict__ bias, float* __restrict__ C,
                    int M, int N, int K)
{
  const int lane   = threadIdx.x & 31;
  const int wave   = threadIdx.x >> 5;
  const int wm     = wave >> 2;   // 0..1
  const int wn     = wave & 3;    // 0..3
  const int laneLo = lane & 15;
  const int laneHi = lane >> 4;
  const int m0 = blockIdx.y * 128 + wm * 64;
  const int n0 = blockIdx.x * 64 + wn * 16;
  if (m0 >= M) return;  // uniform per wave

  const __bf16* ap0 = A + (size_t)(m0 +  0 + laneLo) * K + laneHi * 8;
  const __bf16* ap1 = A + (size_t)(m0 + 16 + laneLo) * K + laneHi * 8;
  const __bf16* ap2 = A + (size_t)(m0 + 32 + laneLo) * K + laneHi * 8;
  const __bf16* ap3 = A + (size_t)(m0 + 48 + laneLo) * K + laneHi * 8;
  const __bf16* bp  = Bt + (size_t)(n0 + laneLo) * K + laneHi * 8;

  v8f acc0 = {}, acc1 = {}, acc2 = {}, acc3 = {};

  auto load_frags = [&](int k0) {
    Frags f;
    f.b  = ld16bf_split8(bp  + k0);
    f.a0 = ld16bf_split8(ap0 + k0);
    f.a1 = ld16bf_split8(ap1 + k0);
    f.a2 = ld16bf_split8(ap2 + k0);
    f.a3 = ld16bf_split8(ap3 + k0);
    return f;
  };

  Frags cur = load_frags(0);
  int k0 = 0;
  for (; k0 + 32 < K; k0 += 32) {
    Frags nxt = load_frags(k0 + 32);                // issue next loads first
    __builtin_prefetch(bp + k0 + 544, 0, 3);        // weight stream, WGP-near scope
    acc0 = __builtin_amdgcn_wmma_f32_16x16x32_bf16(false, cur.a0, false, cur.b, (short)0, acc0, false, false);
    acc1 = __builtin_amdgcn_wmma_f32_16x16x32_bf16(false, cur.a1, false, cur.b, (short)0, acc1, false, false);
    acc2 = __builtin_amdgcn_wmma_f32_16x16x32_bf16(false, cur.a2, false, cur.b, (short)0, acc2, false, false);
    acc3 = __builtin_amdgcn_wmma_f32_16x16x32_bf16(false, cur.a3, false, cur.b, (short)0, acc3, false, false);
    cur = nxt;
  }
  acc0 = __builtin_amdgcn_wmma_f32_16x16x32_bf16(false, cur.a0, false, cur.b, (short)0, acc0, false, false);
  acc1 = __builtin_amdgcn_wmma_f32_16x16x32_bf16(false, cur.a1, false, cur.b, (short)0, acc1, false, false);
  acc2 = __builtin_amdgcn_wmma_f32_16x16x32_bf16(false, cur.a2, false, cur.b, (short)0, acc2, false, false);
  acc3 = __builtin_amdgcn_wmma_f32_16x16x32_bf16(false, cur.a3, false, cur.b, (short)0, acc3, false, false);

  float bv = bias ? bias[n0 + laneLo] : 0.f;
  // C/D layout: VGPR r holds row M = r + 8*(lane>=16); column N = lane%16.
  #pragma unroll
  for (int r = 0; r < 8; ++r) {
    int rr = r + 8 * laneHi;
    C[(size_t)(m0 +  0 + rr) * N + n0 + laneLo] = acc0[r] + bv;
    C[(size_t)(m0 + 16 + rr) * N + n0 + laneLo] = acc1[r] + bv;
    C[(size_t)(m0 + 32 + rr) * N + n0 + laneLo] = acc2[r] + bv;
    C[(size_t)(m0 + 48 + rr) * N + n0 + laneLo] = acc3[r] + bv;
  }
}

// dst[(n)*ldDst + colOff + k] = (bf16) src[k*N + n]   (src is K x N fp32 row-major)
__global__ void conv_tr_kernel(__bf16* __restrict__ dst, const float* __restrict__ src,
                               int K, int N, int ldDst, int colOff)
{
  int i = blockIdx.x * 256 + threadIdx.x;
  if (i >= K * N) return;
  int k = i / N, n = i - k * N;
  dst[(size_t)n * ldDst + colOff + k] = (__bf16)src[i];
}

__global__ void conv_flat_kernel(__bf16* __restrict__ dst, const float* __restrict__ src, int n)
{
  int i = blockIdx.x * 256 + threadIdx.x;
  if (i < n) dst[i] = (__bf16)src[i];
}

__global__ void fzero_kernel(float* __restrict__ p, int n)
{
  int i = blockIdx.x * 256 + threadIdx.x;
  if (i < n) p[i] = 0.f;
}

// Build concatenated bf16 input [a | zb*b | zc*c], row per batch (B=128, rows of 1024 each).
__global__ void prep_cat_kernel(__bf16* __restrict__ out, int width,
                                const float* __restrict__ a, const float* __restrict__ za,
                                const float* __restrict__ b, const float* __restrict__ zb,
                                const float* __restrict__ c, const float* __restrict__ zc)
{
  int i = blockIdx.x * 256 + threadIdx.x;
  if (i >= 128 * width) return;
  int bb = i / width, col = i - bb * width;
  const float* src; const float* zs; int j;
  if (col < 1024)      { src = a; zs = za; j = col; }
  else if (col < 2048) { src = b; zs = zb; j = col - 1024; }
  else                 { src = c; zs = zc; j = col - 2048; }
  float v = src[(size_t)bb * 1024 + j];
  if (zs) v *= zs[bb];
  out[(size_t)bb * width + col] = (__bf16)v;
}

// HMLSTM cell for layers 0/1. One block per batch row. Computes the z column dot
// (4097th gate column) cooperatively, then updates h/c/z with copy + time masking.
__global__ __launch_bounds__(256)
void cell01_kernel(const float* __restrict__ pre, const __bf16* __restrict__ incat,
                   const __bf16* __restrict__ wtz, const float* __restrict__ bz,
                   float* __restrict__ h, float* __restrict__ c,
                   float* __restrict__ zself, const float* __restrict__ zlm1,
                   const int* __restrict__ length, int t)
{
  int b = blockIdx.x;
  __shared__ float red[8];
  const __bf16* row = incat + (size_t)b * 3072;
  float s = 0.f;
  for (int k = threadIdx.x; k < 3072; k += 256)
    s += (float)row[k] * (float)wtz[k];
  for (int o = 16; o > 0; o >>= 1) s += __shfl_down(s, o, 32);
  if ((threadIdx.x & 31) == 0) red[threadIdx.x >> 5] = s;
  __syncthreads();
  float zp = 0.f;
  #pragma unroll
  for (int r = 0; r < 8; ++r) zp += red[r];
  zp += *bz;

  float z_tm1 = zself[b];
  float zl    = zlm1 ? zlm1[b] : 1.0f;
  float keep  = 1.0f - z_tm1;
  float cp    = (1.0f - z_tm1) * (1.0f - zl);
  float tm    = (t < length[b]) ? 1.0f : 0.0f;

  const float* pr = pre + (size_t)b * 4096;
  float* hr = h + (size_t)b * 1024;
  float* cr = c + (size_t)b * 1024;
  for (int hh = threadIdx.x; hh < 1024; hh += 256) {
    float f  = pr[hh];
    float ii = pr[1024 + hh];
    float o  = pr[2048 + hh];
    float g  = pr[3072 + hh];
    float c0 = cr[hh], h0 = hr[hh];
    float c1 = keep * sigf(f) * c0 + sigf(ii) * tanhf(g);
    float h1 = sigf(o) * tanhf(c1);
    float hn = cp * h0 + (1.0f - cp) * h1;
    float cn = cp * c0 + (1.0f - cp) * c1;
    hn = tm * hn + (1.0f - tm) * h0;
    cn = tm * cn + (1.0f - tm) * c0;
    hr[hh] = hn; cr[hh] = cn;
  }
  __syncthreads();  // everyone has read z_tm1 before we overwrite it
  if (threadIdx.x == 0) {
    float hs = fminf(fmaxf((zp + 1.0f) * 0.5f, 0.0f), 1.0f);
    zself[b] = roundf(hs);  // straight-through forward value
  }
}

// Plain-LSTM top layer update (runs AFTER output mixing has consumed old h2).
__global__ void cell2_kernel(const float* __restrict__ pre, float* __restrict__ h2,
                             float* __restrict__ c2, const float* __restrict__ z0,
                             const int* __restrict__ length, int t)
{
  int i = blockIdx.x * 256 + threadIdx.x;   // 0..131071
  int b = i >> 10, hh = i & 1023;
  const float* pr = pre + (size_t)b * 4096;
  float f  = pr[hh];
  float ii = pr[1024 + hh];
  float o  = pr[2048 + hh];
  float g  = pr[3072 + hh];
  float c0 = c2[i], h0 = h2[i];
  float c1 = sigf(f) * c0 + sigf(ii) * tanhf(g);
  float h1 = sigf(o) * tanhf(c1);
  float cp = 1.0f - z0[b];                  // (1-0)*(1-z_lm1)
  float tm = (t < length[b]) ? 1.0f : 0.0f;
  float hn = cp * h0 + (1.0f - cp) * h1;
  float cn = cp * c0 + (1.0f - cp) * c1;
  hn = tm * hn + (1.0f - tm) * h0;
  cn = tm * cn + (1.0f - tm) * c0;
  h2[i] = hn; c2[i] = cn;
}

// gate[bg,lg] = sigmoid( sum_k Ht_flat[bg*3072+k] * Wgate[k,lg] + bgate[lg] ), one wave per output.
__global__ __launch_bounds__(32)
void gate_kernel(float* __restrict__ gateall, const float* __restrict__ hflat,
                 const float* __restrict__ Wgate, const float* __restrict__ bgate)
{
  int j = blockIdx.x;           // 0..383 ; j = bg*3 + lg (== gate.reshape(-1) index)
  int bg = j / 3, lg = j - bg * 3;
  float s = 0.f;
  for (int k = threadIdx.x; k < 3072; k += 32)
    s += hflat[(size_t)bg * 3072 + k] * Wgate[k * 3 + lg];
  for (int o = 16; o > 0; o >>= 1) s += __shfl_down(s, o, 32);
  if (threadIdx.x == 0)
    gateall[j] = 1.0f / (1.0f + expf(-(s + bgate[lg])));
}

// out[b,h] = sum_l gate_flat[l*128+b] * Ht_hat[l*128+b, h]
__global__ void mix_kernel(float* __restrict__ out, const float* __restrict__ hhat,
                           const float* __restrict__ gateall)
{
  int i = blockIdx.x * 256 + threadIdx.x;   // 0..131071 = b*1024 + h
  int b = i >> 10;
  out[i] = gateall[b]       * hhat[i]
         + gateall[128 + b] * hhat[131072 + i]
         + gateall[256 + b] * hhat[262144 + i];
}

extern "C" void kernel_launch(void* const* d_in, const int* in_sizes, int n_in,
                              void* d_out, int out_size, void* d_ws, size_t ws_size,
                              hipStream_t stream)
{
  if (n_in < 17) return;
  const float* input_ = (const float*)d_in[0];
  const int*   length = (const int*)d_in[1];
  const float* Wbh0 = (const float*)d_in[2];
  const float* Whh0 = (const float*)d_in[3];
  const float* Wth0 = (const float*)d_in[4];
  const float* b0   = (const float*)d_in[5];
  const float* Wbh1 = (const float*)d_in[6];
  const float* Whh1 = (const float*)d_in[7];
  const float* Wth1 = (const float*)d_in[8];
  const float* b1   = (const float*)d_in[9];
  const float* Wih2 = (const float*)d_in[10];
  const float* Whh2 = (const float*)d_in[11];
  const float* b2   = (const float*)d_in[12];
  const float* Wout = (const float*)d_in[13];
  const float* bout = (const float*)d_in[14];
  const float* Wgate= (const float*)d_in[15];
  const float* bgate= (const float*)d_in[16];
  float* out = (float*)d_out;

  char* ws = (char*)d_ws;
  size_t off = 0;
  auto alloc = [&](size_t bytes) -> char* {
    char* p = ws + off;
    off = (off + bytes + 255) & ~(size_t)255;
    return p;
  };
  __bf16* wt0    = (__bf16*)alloc((size_t)4097 * 3072 * 2); // [Whh0|Wbh0|Wth0]^T, N-major
  __bf16* wt1    = (__bf16*)alloc((size_t)4097 * 3072 * 2);
  __bf16* wt2    = (__bf16*)alloc((size_t)4096 * 2048 * 2); // [Whh2|Wih2]^T
  __bf16* wtout  = (__bf16*)alloc((size_t)1024 * 1024 * 2); // Wout^T
  __bf16* incat  = (__bf16*)alloc((size_t)128 * 3072 * 2);
  __bf16* hcat   = (__bf16*)alloc((size_t)384 * 1024 * 2);
  float*  pre    = (float*)alloc((size_t)128 * 4096 * 4);
  float*  hhat   = (float*)alloc((size_t)384 * 1024 * 4);
  float*  gateall= (float*)alloc(384 * 4);
  float*  hstate = (float*)alloc((size_t)3 * 128 * 1024 * 4);
  float*  cstate = (float*)alloc((size_t)3 * 128 * 1024 * 4);
  float*  zstate = (float*)alloc(2 * 128 * 4);
  if (off > ws_size) return;

  const int HBH = 128 * 1024;  // one layer's h/c slab

  // ---- init states ----
  fzero_kernel<<<(3 * HBH + 255) / 256, 256, 0, stream>>>(hstate, 3 * HBH);
  fzero_kernel<<<(3 * HBH + 255) / 256, 256, 0, stream>>>(cstate, 3 * HBH);
  fzero_kernel<<<1, 256, 0, stream>>>(zstate, 256);

  // ---- weight conversion (fp32 -> bf16, transposed/concatenated) ----
  int g0 = (1024 * 4097 + 255) / 256;
  conv_tr_kernel<<<g0, 256, 0, stream>>>(wt0, Whh0, 1024, 4097, 3072, 0);
  conv_tr_kernel<<<g0, 256, 0, stream>>>(wt0, Wbh0, 1024, 4097, 3072, 1024);
  conv_tr_kernel<<<g0, 256, 0, stream>>>(wt0, Wth0, 1024, 4097, 3072, 2048);
  conv_tr_kernel<<<g0, 256, 0, stream>>>(wt1, Whh1, 1024, 4097, 3072, 0);
  conv_tr_kernel<<<g0, 256, 0, stream>>>(wt1, Wbh1, 1024, 4097, 3072, 1024);
  conv_tr_kernel<<<g0, 256, 0, stream>>>(wt1, Wth1, 1024, 4097, 3072, 2048);
  int g2 = (1024 * 4096 + 255) / 256;
  conv_tr_kernel<<<g2, 256, 0, stream>>>(wt2, Whh2, 1024, 4096, 2048, 0);
  conv_tr_kernel<<<g2, 256, 0, stream>>>(wt2, Wih2, 1024, 4096, 2048, 1024);
  int go = (1024 * 1024 + 255) / 256;
  conv_tr_kernel<<<go, 256, 0, stream>>>(wtout, Wout, 1024, 1024, 1024, 0);

  dim3 gpre(64, 1);   // N=4096/64, M=128/128
  dim3 gout(16, 3);   // N=1024/64, M=384/128
  const int pc3 = (128 * 3072) / 256;
  const int pc2 = (128 * 2048) / 256;

  for (int t = 0; t < 128; ++t) {
    const float* xt = input_ + (size_t)t * 128 * 1024;

    // ---- layer 0: in = [h0 | x_t | z0_prev*h1] ----
    prep_cat_kernel<<<pc3, 256, 0, stream>>>(incat, 3072,
        hstate, nullptr, xt, nullptr, hstate + HBH, zstate);
    gemm_bf16_wmma<<<gpre, 256, 0, stream>>>(incat, wt0, b0, pre, 128, 4096, 3072);
    cell01_kernel<<<128, 256, 0, stream>>>(pre, incat, wt0 + (size_t)4096 * 3072,
        b0 + 4096, hstate, cstate, zstate, nullptr, length, t);

    // ---- layer 1: in = [h1 | z0_cur*h0_cur | z1_prev*h2] ----
    prep_cat_kernel<<<pc3, 256, 0, stream>>>(incat, 3072,
        hstate + HBH, nullptr, hstate, zstate, hstate + 2 * HBH, zstate + 128);
    gemm_bf16_wmma<<<gpre, 256, 0, stream>>>(incat, wt1, b1, pre, 128, 4096, 3072);
    cell01_kernel<<<128, 256, 0, stream>>>(pre, incat, wt1 + (size_t)4096 * 3072,
        b1 + 4096, hstate + HBH, cstate + HBH, zstate + 128, zstate, length, t);

    // ---- layer 2 pre-activations: in = [h2_prev | h1_cur] ----
    prep_cat_kernel<<<pc2, 256, 0, stream>>>(incat, 2048,
        hstate + 2 * HBH, nullptr, hstate + HBH, nullptr, nullptr, nullptr);
    gemm_bf16_wmma<<<gpre, 256, 0, stream>>>(incat, wt2, b2, pre, 128, 4096, 2048);

    // ---- output mixing (uses h0_cur, h1_cur, h2_prev) ----
    conv_flat_kernel<<<(3 * HBH + 255) / 256, 256, 0, stream>>>(hcat, hstate, 3 * HBH);
    gemm_bf16_wmma<<<gout, 256, 0, stream>>>(hcat, wtout, bout, hhat, 384, 1024, 1024);
    gate_kernel<<<384, 32, 0, stream>>>(gateall, hstate, Wgate, bgate);
    mix_kernel<<<512, 256, 0, stream>>>(out + (size_t)t * HBH, hhat, gateall);

    // ---- commit layer 2 state ----
    cell2_kernel<<<512, 256, 0, stream>>>(pre, hstate + 2 * HBH, cstate + 2 * HBH,
        zstate, length, t);
  }
}
